// GCNFN_52716428591488
// MI455X (gfx1250) — matrix operand order repaired and checked by reading, not compiled, not run on hardware
//
#include <hip/hip_runtime.h>

#define NN   100000
#define EE   1600000
#define ETOT (2*EE + NN)   // 3,300,000
#define GG   128
#define HID  128
#define FIN  256
#define NHID 64
#define NC   4
#define NEG_SLOPE 0.2f
#define LDA  48            // padded LDS row stride (elements): 96B keeps 16B align, spreads banks

typedef __attribute__((ext_vector_type(16))) __bf16 v16bf;
typedef __attribute__((ext_vector_type(8)))  __bf16 bf16x8;
typedef __attribute__((ext_vector_type(4)))  __bf16 bf16x4;
typedef __attribute__((ext_vector_type(8)))  float  v8f;

__device__ __forceinline__ float selu_f(float x) {
    const float A = 1.6732632423543772f, S = 1.0507009873554805f;
    return x > 0.f ? S * x : S * A * (__expf(x) - 1.f);
}
// order-preserving float->uint encoding for atomicMax-based segment_max
__device__ __forceinline__ unsigned enc_f(float f) {
    unsigned u = __float_as_uint(f);
    return (u >> 31) ? ~u : (u | 0x80000000u);
}
__device__ __forceinline__ float dec_f(unsigned k) {
    return __uint_as_float((k >> 31) ? (k & 0x7FFFFFFFu) : ~k);
}
// edge id -> (src,dst) for concat(e, reversed(e), self_loops)
__device__ __forceinline__ void edge_sd(int t, const long long* __restrict__ ei, int& s, int& d) {
    if (t < EE)            { s = (int)ei[t];        d = (int)ei[EE + t]; }
    else if (t < 2 * EE)   { int u = t - EE; s = (int)ei[EE + u]; d = (int)ei[u]; }
    else                   { s = d = t - 2 * EE; }
}

__global__ void k_zero(unsigned* p, int n) {
    int i = blockIdx.x * blockDim.x + threadIdx.x;
    if (i < n) p[i] = 0u;
}

__global__ void k_cvt_bf16(const float* __restrict__ s, __bf16* __restrict__ d, int n) {
    int i = blockIdx.x * blockDim.x + threadIdx.x;
    if (i < n) d[i] = (__bf16)s[i];
}

// C[M,OUT] = A[M,K] @ W[OUT,K]^T  (M mult of 32; K mult of 32; OUT mult of 16, <=128)
// Block: 256 threads = 8 waves. Block owns 32 rows; wave w owns cols [16w,16w+16)
// and computes two 16x16 tiles sharing one B fragment. Double-buffered LDS staging,
// software-pipelined: next K-chunk global load issues before this step's WMMAs.
template <int K, int OUT>
__global__ void k_gemm_bf16(const float* __restrict__ A, const __bf16* __restrict__ W,
                            float* __restrict__ C) {
    __shared__ __bf16 lds_a[2][32 * LDA];
    const int lane = threadIdx.x & 31;
    const int wave = threadIdx.x >> 5;
    const int m0 = blockIdx.x * 32;
    const int n0 = wave * 16;
    const int mr = lane & 15;     // row (A) / col (B) within tile
    const int hi = lane >> 4;     // lane half selects K sub-chunk
    // staging coords: each thread converts 4 consecutive f32 of the 32x32 tile
    const int se = threadIdx.x * 4;
    const int sr = se >> 5;           // 0..31
    const int sc = se & 31;           // 0,4,...,28
    const float* arow = A + (size_t)(m0 + sr) * K + sc;
    v8f acc0 = {}, acc1 = {};

    // prologue: stage K-chunk 0 into buffer 0
    {
        float4 v = *(const float4*)arow;
        bf16x4 t; t[0] = (__bf16)v.x; t[1] = (__bf16)v.y; t[2] = (__bf16)v.z; t[3] = (__bf16)v.w;
        *(bf16x4*)&lds_a[0][sr * LDA + sc] = t;
    }
    __syncthreads();

    constexpr int NSTEPS = K >> 5;
    #pragma unroll
    for (int it = 0; it < NSTEPS; ++it) {
        const int p = it & 1;
        const int k0 = it << 5;
        // issue next-chunk global load early so its latency hides under the WMMAs
        float4 vnext;
        if (it + 1 < NSTEPS)
            vnext = *(const float4*)&arow[k0 + 32];
        if (OUT == 128 || n0 < OUT) {   // compile-time folds for OUT==128
            union { bf16x8 h[2]; v16bf v; } a0, a1;
            // A k order used here (contiguous, not ISA table order): lane half hi: K = hi*16 + [0,16)
            a0.h[0] = *(const bf16x8*)&lds_a[p][mr * LDA + hi * 16];
            a0.h[1] = *(const bf16x8*)&lds_a[p][mr * LDA + hi * 16 + 8];
            a1.h[0] = *(const bf16x8*)&lds_a[p][(16 + mr) * LDA + hi * 16];
            a1.h[1] = *(const bf16x8*)&lds_a[p][(16 + mr) * LDA + hi * 16 + 8];
            // B layout: lane holds col n=mr, K = hi*16 + [0,16)
            v16bf bf = *(const v16bf*)&W[(size_t)(n0 + mr) * K + k0 + hi * 16];
            acc0 = __builtin_amdgcn_wmma_f32_16x16x32_bf16(
                false, a0.v, false, bf, (short)0, acc0, false, false);
            acc1 = __builtin_amdgcn_wmma_f32_16x16x32_bf16(
                false, a1.v, false, bf, (short)0, acc1, false, false);
        }
        if (it + 1 < NSTEPS) {   // convert + store next chunk into other buffer
            bf16x4 t;
            t[0] = (__bf16)vnext.x; t[1] = (__bf16)vnext.y;
            t[2] = (__bf16)vnext.z; t[3] = (__bf16)vnext.w;
            *(bf16x4*)&lds_a[1 - p][sr * LDA + sc] = t;
        }
        __syncthreads();
    }
    if (OUT == 128 || n0 < OUT) {
        #pragma unroll
        for (int r = 0; r < 8; r++) {  // C/D: M = hi*8 + r, N = mr
            C[(size_t)(m0 + hi * 8 + r) * OUT + n0 + mr]      = acc0[r];
            C[(size_t)(m0 + 16 + hi * 8 + r) * OUT + n0 + mr] = acc1[r];
        }
    }
}

// per-node attention scores: ssrc = H.a_src, sdst = H.a_dst (wave per node)
__global__ void k_scores(const float* __restrict__ H, const float* __restrict__ avs,
                         const float* __restrict__ avd, float* ssrc, float* sdst) {
    int w = (blockIdx.x * blockDim.x + threadIdx.x) >> 5;
    if (w >= NN) return;
    int lane = threadIdx.x & 31;
    const float* row = H + (size_t)w * HID;
    float s1 = 0.f, s2 = 0.f;
    #pragma unroll
    for (int j = 0; j < HID / 32; j++) {
        float h = row[lane + 32 * j];
        s1 += h * avs[lane + 32 * j];
        s2 += h * avd[lane + 32 * j];
    }
    #pragma unroll
    for (int m = 16; m >= 1; m >>= 1) {
        s1 += __shfl_xor(s1, m, 32);
        s2 += __shfl_xor(s2, m, 32);
    }
    if (lane == 0) { ssrc[w] = s1; sdst[w] = s2; }
}

__global__ void k_edge_max(const long long* __restrict__ ei, const float* __restrict__ ssrc,
                           const float* __restrict__ sdst, unsigned* __restrict__ menc) {
    int t = blockIdx.x * blockDim.x + threadIdx.x;
    if (t >= ETOT) return;
    int s, d; edge_sd(t, ei, s, d);
    float e = ssrc[s] + sdst[d];
    e = e > 0.f ? e : NEG_SLOPE * e;
    atomicMax(menc + d, enc_f(e));
}

__global__ void k_edge_expsum(const long long* __restrict__ ei, const float* __restrict__ ssrc,
                              const float* __restrict__ sdst, const unsigned* __restrict__ menc,
                              float* __restrict__ den, float* __restrict__ wedge) {
    int t = blockIdx.x * blockDim.x + threadIdx.x;
    if (t >= ETOT) return;
    int s, d; edge_sd(t, ei, s, d);
    float e = ssrc[s] + sdst[d];
    e = e > 0.f ? e : NEG_SLOPE * e;
    float w = __expf(e - dec_f(menc[d]));
    wedge[t] = w;
    atomicAdd(den + d, w);
}

// wave per edge: out[dst] += alpha * h[src]   (4 feats per lane)
__global__ void k_edge_agg(const long long* __restrict__ ei, const float* __restrict__ wedge,
                           const float* __restrict__ den, const float* __restrict__ H,
                           float* __restrict__ OUTA) {
    int w = (int)(((size_t)blockIdx.x * blockDim.x + threadIdx.x) >> 5);
    if (w >= ETOT) return;
    int lane = threadIdx.x & 31;
    int s, d; edge_sd(w, ei, s, d);
    float alpha = wedge[w] / den[d];
    float4 v = ((const float4*)(H + (size_t)s * HID))[lane];
    float* od = OUTA + (size_t)d * HID + lane * 4;
    atomicAdd(od + 0, v.x * alpha);
    atomicAdd(od + 1, v.y * alpha);
    atomicAdd(od + 2, v.z * alpha);
    atomicAdd(od + 3, v.w * alpha);
}

__global__ void k_bias_selu(float* h, const float* __restrict__ b, int n, int width) {
    int i = blockIdx.x * blockDim.x + threadIdx.x;
    if (i < n) h[i] = selu_f(h[i] + b[i & (width - 1)]);
}

// global mean-pool: wave per node
__global__ void k_pool(const float* __restrict__ h, const long long* __restrict__ batch,
                       float* __restrict__ gsum, float* __restrict__ cnt) {
    int w = (blockIdx.x * blockDim.x + threadIdx.x) >> 5;
    if (w >= NN) return;
    int lane = threadIdx.x & 31;
    int g = (int)batch[w];
    float4 v = ((const float4*)(h + (size_t)w * HID))[lane];
    float* gs = gsum + g * HID + lane * 4;
    atomicAdd(gs + 0, v.x); atomicAdd(gs + 1, v.y);
    atomicAdd(gs + 2, v.z); atomicAdd(gs + 3, v.w);
    if (lane == 0) atomicAdd(cnt + g, 1.f);
}

__global__ void k_graph(const float* __restrict__ gsum, const float* __restrict__ cnt,
                        float* __restrict__ gbuf) {
    int i = blockIdx.x * blockDim.x + threadIdx.x;
    if (i >= GG * HID) return;
    gbuf[i] = selu_f(gsum[i] / fmaxf(cnt[i >> 7], 1.f));
}

__global__ void k_fc2(const float* __restrict__ gsel, const float* __restrict__ fw2,
                      const float* __restrict__ fb2, float* __restrict__ out) {
    int g = blockIdx.x * blockDim.x + threadIdx.x;
    if (g >= GG) return;
    const float* row = gsel + g * NHID;
    float lg[NC]; float mx = -1e30f;
    #pragma unroll
    for (int c = 0; c < NC; c++) {
        float s = fb2[c];
        for (int j = 0; j < NHID; j++) s += row[j] * fw2[c * NHID + j];
        lg[c] = s; mx = fmaxf(mx, s);
    }
    float se = 0.f;
    #pragma unroll
    for (int c = 0; c < NC; c++) se += __expf(lg[c] - mx);
    float lse = mx + __logf(se);
    #pragma unroll
    for (int c = 0; c < NC; c++) out[g * NC + c] = lg[c] - lse;
}

#define CDIV(a, b) (((a) + (b) - 1) / (b))

extern "C" void kernel_launch(void* const* d_in, const int* in_sizes, int n_in,
                              void* d_out, int out_size, void* d_ws, size_t ws_size,
                              hipStream_t stream) {
    (void)in_sizes; (void)n_in; (void)out_size; (void)ws_size;
    const float*     x     = (const float*)d_in[0];
    const long long* ei    = (const long long*)d_in[1];
    const long long* batch = (const long long*)d_in[2];
    const float* W1  = (const float*)d_in[3];
    const float* as1 = (const float*)d_in[4];
    const float* ad1 = (const float*)d_in[5];
    const float* b1  = (const float*)d_in[6];
    const float* W2  = (const float*)d_in[7];
    const float* as2 = (const float*)d_in[8];
    const float* ad2 = (const float*)d_in[9];
    const float* b2  = (const float*)d_in[10];
    const float* fw1 = (const float*)d_in[11];
    const float* fb1 = (const float*)d_in[12];
    const float* fw2 = (const float*)d_in[13];
    const float* fb2 = (const float*)d_in[14];

    char* wp = (char*)d_ws;
    auto take = [&](size_t bytes) -> void* {
        void* p = (void*)wp;
        wp += (bytes + 255) & ~(size_t)255;
        return p;
    };
    float*    H     = (float*)take((size_t)NN * HID * 4);
    float*    OUTA  = (float*)take((size_t)NN * HID * 4);
    float*    ssrc  = (float*)take((size_t)NN * 4);
    float*    sdst  = (float*)take((size_t)NN * 4);
    unsigned* menc  = (unsigned*)take((size_t)NN * 4);
    float*    den   = (float*)take((size_t)NN * 4);
    float*    wedge = (float*)take((size_t)ETOT * 4);
    __bf16*   W1b   = (__bf16*)take((size_t)HID * FIN * 2);
    __bf16*   W2b   = (__bf16*)take((size_t)HID * HID * 2);
    __bf16*   fw1b  = (__bf16*)take((size_t)NHID * HID * 2);
    float*    gsum  = (float*)take((size_t)GG * HID * 4);
    float*    cnt   = (float*)take((size_t)GG * 4);
    float*    gbuf  = (float*)take((size_t)GG * HID * 4);
    float*    z1    = (float*)take((size_t)GG * NHID * 4);

    const int T = 256;
    // pre-convert weights to bf16 once per call
    k_cvt_bf16<<<CDIV(HID * FIN, T), T, 0, stream>>>(W1, W1b, HID * FIN);
    k_cvt_bf16<<<CDIV(HID * HID, T), T, 0, stream>>>(W2, W2b, HID * HID);
    k_cvt_bf16<<<CDIV(NHID * HID, T), T, 0, stream>>>(fw1, fw1b, NHID * HID);

    // ---- GAT layer 1 ----
    k_gemm_bf16<FIN, HID><<<NN / 32, T, 0, stream>>>(x, W1b, H);
    k_scores<<<NN * 32 / T, T, 0, stream>>>(H, as1, ad1, ssrc, sdst);
    k_zero<<<CDIV(NN, T), T, 0, stream>>>(menc, NN);
    k_zero<<<CDIV(NN, T), T, 0, stream>>>((unsigned*)den, NN);
    k_zero<<<CDIV(NN * HID, T), T, 0, stream>>>((unsigned*)OUTA, NN * HID);
    k_edge_max<<<CDIV(ETOT, T), T, 0, stream>>>(ei, ssrc, sdst, menc);
    k_edge_expsum<<<CDIV(ETOT, T), T, 0, stream>>>(ei, ssrc, sdst, menc, den, wedge);
    k_edge_agg<<<ETOT / 8, T, 0, stream>>>(ei, wedge, den, H, OUTA);
    k_bias_selu<<<CDIV(NN * HID, T), T, 0, stream>>>(OUTA, b1, NN * HID, HID);

    // ---- GAT layer 2 (H buffer reused; OUTA consumed by gemm, then reused) ----
    k_gemm_bf16<HID, HID><<<NN / 32, T, 0, stream>>>(OUTA, W2b, H);
    k_scores<<<NN * 32 / T, T, 0, stream>>>(H, as2, ad2, ssrc, sdst);
    k_zero<<<CDIV(NN, T), T, 0, stream>>>(menc, NN);
    k_zero<<<CDIV(NN, T), T, 0, stream>>>((unsigned*)den, NN);
    k_zero<<<CDIV(NN * HID, T), T, 0, stream>>>((unsigned*)OUTA, NN * HID);
    k_edge_max<<<CDIV(ETOT, T), T, 0, stream>>>(ei, ssrc, sdst, menc);
    k_edge_expsum<<<CDIV(ETOT, T), T, 0, stream>>>(ei, ssrc, sdst, menc, den, wedge);
    k_edge_agg<<<ETOT / 8, T, 0, stream>>>(ei, wedge, den, H, OUTA);
    k_bias_selu<<<CDIV(NN * HID, T), T, 0, stream>>>(OUTA, b2, NN * HID, HID);

    // ---- mean pool + head ----
    k_zero<<<CDIV(GG * HID, T), T, 0, stream>>>((unsigned*)gsum, GG * HID);
    k_zero<<<1, T, 0, stream>>>((unsigned*)cnt, GG);
    k_pool<<<NN * 32 / T, T, 0, stream>>>(OUTA, batch, gsum, cnt);
    k_graph<<<CDIV(GG * HID, T), T, 0, stream>>>(gsum, cnt, gbuf);
    k_gemm_bf16<HID, NHID><<<GG / 32, T, 0, stream>>>(gbuf, fw1b, z1);
    k_bias_selu<<<CDIV(GG * NHID, T), T, 0, stream>>>(z1, fb1, GG * NHID, NHID);
    k_fc2<<<1, 128, 0, stream>>>(z1, fw2, fb2, (float*)d_out);
}
